// StarFusionBottleneck_V3_78520592105641
// MI455X (gfx1250) — compile-verified, matrix operand order checked by reading, not run on hardware
//
#include <hip/hip_runtime.h>
#include <hip/hip_bf16.h>
#include <math.h>

// ---------------------------------------------------------------------------
// StarFusionBottleneck for MI455X (gfx1250, wave32).
// Memory-bound op (~210MB compulsory HBM @23.3TB/s ≈ 9us floor, ~15GF total),
// so: fp16 intermediates (L2-resident, 192MB), WMMA f16 for the three channel
// GEMMs (reduce 256->128, dcn pointwise 64->64, expand 128->256), VALU for the
// depthwise / gather / attention branches. Needs ~107MB of d_ws scratch.
// GEMM inner loops batch all LDS fragment loads ahead of a run of 8 (or 4)
// back-to-back v_wmma ops on disjoint accumulators (no inter-WMMA hazards),
// instead of load->wait->wmma serialization.
// ---------------------------------------------------------------------------

typedef _Float16 half_t;
typedef __attribute__((ext_vector_type(16))) _Float16 v16h;
typedef __attribute__((ext_vector_type(8)))  _Float16 v8h;
typedef __attribute__((ext_vector_type(8)))  float    v8f;

#define HWN 6400
#define WD  80
#define HD  80
#define NBATCH 16

static __device__ __forceinline__ int imin(int a, int b) { return a < b ? a : b; }
static __device__ __forceinline__ int imax(int a, int b) { return a > b ? a : b; }
static __device__ __forceinline__ float siluf(float v) { return v / (1.f + __expf(-v)); }
static __device__ __forceinline__ float sigmf(float v) { return 1.f / (1.f + __expf(-v)); }
// bn params stacked [4][C] = {gamma, beta, mean, var}
static __device__ __forceinline__ float bnap(const float* p, int C, int c, float v) {
  float s = p[c] * rsqrtf(p[3 * C + c] + 1e-5f);
  return v * s + p[C + c] - p[2 * C + c] * s;
}

// Load one 16-halfword A/B fragment for v_wmma_f32_16x16x32_f16 from LDS.
// ISA 16-bit A layout: lanes 0-15 hold K=koff..koff+7 (halves 0-7) and
// K=koff+16..koff+23 (halves 8-15); lanes 16-31 get koff+8 (caller adds it).
// pitch=72 halves => row stride 36 dwords => fragment b128 reads tile all 64
// LDS banks conflict-free.
static __device__ __forceinline__ v16h load_frag(const half_t* s, int row, int koff, int pitch) {
  const half_t* p = s + row * pitch + koff;
  v8h lo = *(const v8h*)(p);
  v8h hi = *(const v8h*)(p + 16);
  v16h r;
#pragma unroll
  for (int i = 0; i < 8; i++) { r[i] = lo[i]; r[i + 8] = hi[i]; }
  return r;
}

static __device__ __forceinline__ v8f wmma_f16(v16h a, v16h b, v8f c) {
  return __builtin_amdgcn_wmma_f32_16x16x32_f16(false, a, false, b, (short)0, c,
                                                false, false);
}

// ---------------------------------------------------------------------------
// Kernel 1: reduce conv 1x1 (GEMM M=128px tile, N=128, K=256) + BN + SiLU.
// Writes h (fp16, [b][128][hw]) with LDS transpose for coalesced stores.
// ---------------------------------------------------------------------------
__global__ __launch_bounds__(256) void k_reduce(const float* __restrict__ x,
                                                const float* __restrict__ w,
                                                const float* __restrict__ bnp,
                                                half_t* __restrict__ h) {
  __shared__ alignas(16) half_t smem[2 * 128 * 72];
  half_t* sA = smem;
  half_t* sB = smem + 128 * 72;
  const int tile = blockIdx.x % 50;
  const int b = blockIdx.x / 50;
  const int hw0 = tile * 128;
  const int tid = threadIdx.x;
  const int lane = tid & 31;
  const int wv = tid >> 5;      // 0..7 -> M strip
  const int mrow = lane & 15;
  const int hi = lane >> 4;     // K-half select

  v8f acc[8];
#pragma unroll
  for (int i = 0; i < 8; i++)
#pragma unroll
    for (int j = 0; j < 8; j++) acc[i][j] = 0.f;

  const float* xb = x + (size_t)b * 256 * HWN + hw0;

  for (int kt = 0; kt < 256; kt += 64) {
    // stage A: 128 pixels x 64 input channels (coalesced over pixels)
    for (int i = tid; i < 128 * 64; i += 256) {
      int c = i >> 7, m = i & 127;
      sA[m * 72 + c] = (half_t)xb[(size_t)(kt + c) * HWN + m];
      if (kt + 64 < 256)
        __builtin_prefetch(&xb[(size_t)(kt + 64 + c) * HWN + m], 0, 1);
    }
    // stage B: 128 out-channels x 64 k (w_reduce is [128][256])
    for (int i = tid; i < 128 * 64; i += 256) {
      int n = i >> 6, kk = i & 63;
      sB[n * 72 + kk] = (half_t)w[n * 256 + kt + kk];
    }
    __syncthreads();
#pragma unroll
    for (int kk = 0; kk < 64; kk += 32) {
      int koff = kk + hi * 8;
      // Batch all fragment loads, then 8 hazard-free back-to-back WMMAs.
      v16h a = load_frag(sA, wv * 16 + mrow, koff, 72);
      v16h bfr[8];
#pragma unroll
      for (int ns = 0; ns < 8; ns++)
        bfr[ns] = load_frag(sB, ns * 16 + mrow, koff, 72);
#pragma unroll
      for (int ns = 0; ns < 8; ns++)
        acc[ns] = wmma_f16(a, bfr[ns], acc[ns]);
    }
    __syncthreads();
  }

  // BN + SiLU epilogue, transpose through LDS for coalesced fp16 stores.
#pragma unroll
  for (int ns = 0; ns < 8; ns++) {
    int n = ns * 16 + mrow;
    float s = bnp[n] * rsqrtf(bnp[384 + n] + 1e-5f);
    float bi = bnp[128 + n] - bnp[256 + n] * s;
#pragma unroll
    for (int r = 0; r < 8; r++) {
      int m = wv * 16 + r + 8 * hi;
      smem[n * 136 + m] = (half_t)siluf(acc[ns][r] * s + bi);
    }
  }
  __syncthreads();
  for (int i = tid; i < 128 * 128; i += 256) {
    int n = i >> 7, m = i & 127;
    h[((size_t)b * 128 + n) * HWN + hw0 + m] = smem[n * 136 + m];
  }
}

// ---------------------------------------------------------------------------
// Kernel 2: offset conv 3x3 (3 out ch) + sigmoid mask + bilinear grid sample.
// ---------------------------------------------------------------------------
__global__ __launch_bounds__(256) void k_offset(const half_t* __restrict__ h,
                                                const float* __restrict__ offw,
                                                const float* __restrict__ offb,
                                                half_t* __restrict__ xd) {
  __shared__ float sw[3 * 64 * 9];
  for (int i = threadIdx.x; i < 3 * 64 * 9; i += 256) sw[i] = offw[i];
  __syncthreads();
  int p = blockIdx.x * 256 + threadIdx.x;   // exact: 400 blocks
  int b = p / HWN, hw = p % HWN;
  int yy = hw / WD, xx = hw % WD;
  const half_t* hs = h + (size_t)b * 128 * HWN;   // first 64 ch of h
  float o0 = offb[0], o1 = offb[1], o2 = offb[2];
  for (int c = 0; c < 64; c++) {
    const half_t* hc = hs + (size_t)c * HWN;
#pragma unroll
    for (int t = 0; t < 9; t++) {
      int iy = yy + t / 3 - 1, ix = xx + t % 3 - 1;
      float v = 0.f;
      if (iy >= 0 && iy < HD && ix >= 0 && ix < WD) v = (float)hc[iy * WD + ix];
      o0 += sw[0 * 576 + c * 9 + t] * v;
      o1 += sw[1 * 576 + c * 9 + t] * v;
      o2 += sw[2 * 576 + c * 9 + t] * v;
    }
  }
  float mask = sigmf(o2);
  float gxn = -1.f + 2.f * xx / (WD - 1) + o1 * (2.f / WD);
  float gyn = -1.f + 2.f * yy / (HD - 1) + o0 * (2.f / HD);
  float px = (gxn + 1.f) * 0.5f * (WD - 1);
  float py = (gyn + 1.f) * 0.5f * (HD - 1);
  int ix0 = (int)floorf(px), iy0 = (int)floorf(py);
  float fx = px - ix0, fy = py - iy0;
  float vx0 = (ix0 >= 0 && ix0 < WD) ? 1.f : 0.f;
  float vx1 = (ix0 + 1 >= 0 && ix0 + 1 < WD) ? 1.f : 0.f;
  float vy0 = (iy0 >= 0 && iy0 < HD) ? 1.f : 0.f;
  float vy1 = (iy0 + 1 >= 0 && iy0 + 1 < HD) ? 1.f : 0.f;
  int cx0 = imin(imax(ix0, 0), WD - 1), cx1 = imin(imax(ix0 + 1, 0), WD - 1);
  int cy0 = imin(imax(iy0, 0), HD - 1), cy1 = imin(imax(iy0 + 1, 0), HD - 1);
  float w00 = (1 - fx) * (1 - fy) * vx0 * vy0 * mask;
  float w10 = fx * (1 - fy) * vx1 * vy0 * mask;
  float w01 = (1 - fx) * fy * vx0 * vy1 * mask;
  float w11 = fx * fy * vx1 * vy1 * mask;
  int a00 = cy0 * WD + cx0, a10 = cy0 * WD + cx1;
  int a01 = cy1 * WD + cx0, a11 = cy1 * WD + cx1;
  for (int c = 0; c < 64; c++) {
    const half_t* hc = hs + (size_t)c * HWN;
    float v = w00 * (float)hc[a00] + w10 * (float)hc[a10] +
              w01 * (float)hc[a01] + w11 * (float)hc[a11];
    xd[((size_t)b * 64 + c) * HWN + hw] = (half_t)v;
  }
}

// ---------------------------------------------------------------------------
// Kernel 3: DCN depthwise 3x3 pad1 (no BN/act here, matches reference).
// ---------------------------------------------------------------------------
__global__ __launch_bounds__(256) void k_dw(const half_t* __restrict__ xd,
                                            const float* __restrict__ dww,
                                            half_t* __restrict__ t) {
  __shared__ float sw[64 * 9];
  for (int i = threadIdx.x; i < 64 * 9; i += 256) sw[i] = dww[i];
  __syncthreads();
  int idx = blockIdx.x * 256 + threadIdx.x;          // 25600 blocks exact
  int hw = idx % HWN; int bc = idx / HWN; int c = bc & 63;
  int yy = hw / WD, xx = hw % WD;
  const half_t* src = xd + (size_t)bc * HWN;
  float acc = 0.f;
#pragma unroll
  for (int tt = 0; tt < 9; tt++) {
    int iy = yy + tt / 3 - 1, ix = xx + tt % 3 - 1;
    if (iy >= 0 && iy < HD && ix >= 0 && ix < WD)
      acc += sw[c * 9 + tt] * (float)src[iy * WD + ix];
  }
  t[(size_t)bc * HWN + hw] = (half_t)acc;
}

// ---------------------------------------------------------------------------
// Kernel 4: DCN pointwise 64x64 GEMM (WMMA) + BN + SiLU -> dcn (fp16).
// ---------------------------------------------------------------------------
__global__ __launch_bounds__(256) void k_pw(const half_t* __restrict__ t,
                                            const float* __restrict__ w,
                                            const float* __restrict__ bnp,
                                            half_t* __restrict__ dcn) {
  __shared__ alignas(16) half_t smem[128 * 72 + 64 * 72];
  half_t* sA = smem;
  half_t* sB = smem + 128 * 72;
  const int tile = blockIdx.x % 50, b = blockIdx.x / 50;
  const int hw0 = tile * 128;
  const int tid = threadIdx.x, lane = tid & 31, wv = tid >> 5;
  const int mrow = lane & 15, hi = lane >> 4;
  const half_t* tb = t + (size_t)b * 64 * HWN + hw0;
  for (int i = tid; i < 128 * 64; i += 256) {
    int c = i >> 7, m = i & 127;
    sA[m * 72 + c] = tb[(size_t)c * HWN + m];
  }
  for (int i = tid; i < 64 * 64; i += 256) {
    int n = i >> 6, kk = i & 63;
    sB[n * 72 + kk] = (half_t)w[n * 64 + kk];
  }
  __syncthreads();
  v8f acc[4];
#pragma unroll
  for (int i = 0; i < 4; i++)
#pragma unroll
    for (int j = 0; j < 8; j++) acc[i][j] = 0.f;
#pragma unroll
  for (int kk = 0; kk < 64; kk += 32) {
    int koff = kk + hi * 8;
    v16h a = load_frag(sA, wv * 16 + mrow, koff, 72);
    v16h bfr[4];
#pragma unroll
    for (int ns = 0; ns < 4; ns++)
      bfr[ns] = load_frag(sB, ns * 16 + mrow, koff, 72);
#pragma unroll
    for (int ns = 0; ns < 4; ns++)
      acc[ns] = wmma_f16(a, bfr[ns], acc[ns]);
  }
  __syncthreads();
#pragma unroll
  for (int ns = 0; ns < 4; ns++) {
    int n = ns * 16 + mrow;
    float s = bnp[n] * rsqrtf(bnp[192 + n] + 1e-5f);
    float bi = bnp[64 + n] - bnp[128 + n] * s;
#pragma unroll
    for (int r = 0; r < 8; r++) {
      int m = wv * 16 + r + 8 * hi;
      smem[n * 136 + m] = (half_t)siluf(acc[ns][r] * s + bi);
    }
  }
  __syncthreads();
  half_t* ob = dcn + (size_t)b * 64 * HWN + hw0;
  for (int i = tid; i < 64 * 128; i += 256) {
    int n = i >> 7, m = i & 127;
    ob[(size_t)n * HWN + m] = smem[n * 136 + m];
  }
}

// ---------------------------------------------------------------------------
// Kernel 5: asym1 = silu(bn(depthwise 1x7 pad(0,3)))
// ---------------------------------------------------------------------------
__global__ __launch_bounds__(256) void k_asym1(const half_t* __restrict__ h,
                                               const float* __restrict__ wa,
                                               const float* __restrict__ bnp,
                                               half_t* __restrict__ a1) {
  __shared__ float sw[64 * 7];
  for (int i = threadIdx.x; i < 64 * 7; i += 256) sw[i] = wa[i];
  __syncthreads();
  int idx = blockIdx.x * 256 + threadIdx.x;
  int hw = idx % HWN; int bc = idx / HWN; int c = bc & 63; int b = bc >> 6;
  int yy = hw / WD, xx = hw % WD;
  const half_t* src = h + ((size_t)b * 128 + c) * HWN;  // hs channel c
  float acc = 0.f;
#pragma unroll
  for (int k = -3; k <= 3; k++) {
    int ix = xx + k;
    if (ix >= 0 && ix < WD) acc += sw[c * 7 + k + 3] * (float)src[yy * WD + ix];
  }
  a1[(size_t)bc * HWN + hw] = (half_t)siluf(bnap(bnp, 64, c, acc));
}

// ---------------------------------------------------------------------------
// Kernel 6: ctx = softmax(cw) . [d2, d3, asym2]  (all depthwise branches)
// ---------------------------------------------------------------------------
__global__ __launch_bounds__(256) void k_ctx(const half_t* __restrict__ h,
                                             const half_t* __restrict__ a1,
                                             const float* __restrict__ wd2,
                                             const float* __restrict__ bnd2,
                                             const float* __restrict__ wd3,
                                             const float* __restrict__ bnd3,
                                             const float* __restrict__ wa2,
                                             const float* __restrict__ bna2,
                                             const float* __restrict__ cw,
                                             half_t* __restrict__ ctx) {
  __shared__ float s2[64 * 9], s3[64 * 9], sa[64 * 7], scw[3];
  for (int i = threadIdx.x; i < 64 * 9; i += 256) { s2[i] = wd2[i]; s3[i] = wd3[i]; }
  for (int i = threadIdx.x; i < 64 * 7; i += 256) sa[i] = wa2[i];
  if (threadIdx.x < 3) scw[threadIdx.x] = cw[threadIdx.x];
  __syncthreads();
  int idx = blockIdx.x * 256 + threadIdx.x;
  int hw = idx % HWN; int bc = idx / HWN; int c = bc & 63; int b = bc >> 6;
  int yy = hw / WD, xx = hw % WD;
  const half_t* src = h + ((size_t)b * 128 + c) * HWN;  // hs channel c
  float d2 = 0.f, d3 = 0.f;
#pragma unroll
  for (int tt = 0; tt < 9; tt++) {
    int dy = tt / 3 - 1, dx = tt % 3 - 1;
    int iy2 = yy + 2 * dy, ix2 = xx + 2 * dx;
    if (iy2 >= 0 && iy2 < HD && ix2 >= 0 && ix2 < WD)
      d2 += s2[c * 9 + tt] * (float)src[iy2 * WD + ix2];
    int iy3 = yy + 3 * dy, ix3 = xx + 3 * dx;
    if (iy3 >= 0 && iy3 < HD && ix3 >= 0 && ix3 < WD)
      d3 += s3[c * 9 + tt] * (float)src[iy3 * WD + ix3];
  }
  const half_t* a1c = a1 + (size_t)bc * HWN;
  float as = 0.f;
#pragma unroll
  for (int k = -3; k <= 3; k++) {
    int iy = yy + k;
    if (iy >= 0 && iy < HD) as += sa[c * 7 + k + 3] * (float)a1c[iy * WD + xx];
  }
  d2 = siluf(bnap(bnd2, 64, c, d2));
  d3 = siluf(bnap(bnd3, 64, c, d3));
  as = siluf(bnap(bna2, 64, c, as));
  float mx = fmaxf(scw[0], fmaxf(scw[1], scw[2]));
  float e0 = __expf(scw[0] - mx), e1 = __expf(scw[1] - mx), e2 = __expf(scw[2] - mx);
  float inv = 1.f / (e0 + e1 + e2);
  ctx[(size_t)bc * HWN + hw] = (half_t)((e0 * d2 + e1 * d3 + e2 * as) * inv);
}

// ---------------------------------------------------------------------------
// Kernel 7: star0 = bn_star(dcn*ctx) + per-(b,c) channel sums (block reduce).
// ---------------------------------------------------------------------------
__global__ __launch_bounds__(256) void k_star(const half_t* __restrict__ dcn,
                                              const half_t* __restrict__ ctx,
                                              const float* __restrict__ bnst,
                                              half_t* __restrict__ star0,
                                              float* __restrict__ chsum) {
  int bc = blockIdx.x;                 // 0..1023 = b*64+c
  int c = bc & 63;
  int tid = threadIdx.x;
  float s = bnst[c] * rsqrtf(bnst[192 + c] + 1e-5f);
  float bi = bnst[64 + c] - bnst[128 + c] * s;
  float part = 0.f;
  for (int hw = tid; hw < HWN; hw += 256) {
    float v = (float)dcn[(size_t)bc * HWN + hw] * (float)ctx[(size_t)bc * HWN + hw];
    v = v * s + bi;
    star0[(size_t)bc * HWN + hw] = (half_t)v;
    part += v;
  }
  __shared__ float red[256];
  red[tid] = part;
  __syncthreads();
  for (int st = 128; st > 0; st >>= 1) {
    if (tid < st) red[tid] += red[tid + st];
    __syncthreads();
  }
  if (tid == 0) chsum[bc] = red[0];
}

// ---------------------------------------------------------------------------
// Kernel 8: ECA (1D conv k=5 pad2 over channel means) -> sigmoid -> ych.
// ---------------------------------------------------------------------------
__global__ void k_eca(const float* __restrict__ chsum,
                      const float* __restrict__ ew,
                      float* __restrict__ ych) {
  int i = threadIdx.x;          // one 1024-thread block: 16*64
  int b = i >> 6, c = i & 63;
  float acc = 0.f;
#pragma unroll
  for (int t = 0; t < 5; t++) {
    int j = c - 2 + t;
    if (j >= 0 && j < 64) acc += ew[t] * chsum[b * 64 + j] * (1.f / (float)HWN);
  }
  ych[i] = sigmf(acc);
}

// ---------------------------------------------------------------------------
// Kernel 9: spatial mean/max over channels of star0*ych.
// ---------------------------------------------------------------------------
__global__ __launch_bounds__(256) void k_sp(const half_t* __restrict__ star0,
                                            const float* __restrict__ ych,
                                            float* __restrict__ spmean,
                                            float* __restrict__ spmax) {
  int p = blockIdx.x * 256 + threadIdx.x;
  int b = p / HWN, hw = p % HWN;
  float mn = 0.f, mx = -3.4e38f;
  for (int c = 0; c < 64; c++) {
    float v = (float)star0[((size_t)b * 64 + c) * HWN + hw] * ych[b * 64 + c];
    mn += v;
    mx = fmaxf(mx, v);
  }
  spmean[p] = mn * (1.f / 64.f);
  spmax[p] = mx;
}

// ---------------------------------------------------------------------------
// Kernel 10: 7x7 spatial-gate conv over [mean,max] -> sigmoid -> sgate (fp16).
// ---------------------------------------------------------------------------
__global__ __launch_bounds__(256) void k_gate(const float* __restrict__ spmean,
                                              const float* __restrict__ spmax,
                                              const float* __restrict__ spw,
                                              half_t* __restrict__ sgate) {
  __shared__ float sw[98];
  for (int i = threadIdx.x; i < 98; i += 256) sw[i] = spw[i];
  __syncthreads();
  int p = blockIdx.x * 256 + threadIdx.x;
  int b = p / HWN, hw = p % HWN;
  int yy = hw / WD, xx = hw % WD;
  const float* m0 = spmean + (size_t)b * HWN;
  const float* m1 = spmax + (size_t)b * HWN;
  float acc = 0.f;
#pragma unroll
  for (int ky = -3; ky <= 3; ky++) {
    int iy = yy + ky;
    if (iy < 0 || iy >= HD) continue;
#pragma unroll
    for (int kx = -3; kx <= 3; kx++) {
      int ix = xx + kx;
      if (ix < 0 || ix >= WD) continue;
      int t = (ky + 3) * 7 + (kx + 3);
      acc += sw[t] * m0[iy * WD + ix] + sw[49 + t] * m1[iy * WD + ix];
    }
  }
  sgate[p] = (half_t)sigmf(acc);
}

// ---------------------------------------------------------------------------
// Kernel 11: expand conv 1x1 (GEMM N=256 split over blockIdx.y, K=128) with
// on-the-fly blended/shuffled A build, BN + SiLU + residual -> fp32 out.
// shuffled[2i] = alpha*star0[i]*ych[i]*sgate, shuffled[2i+1] = (1-a)*hb[i].
// ---------------------------------------------------------------------------
__global__ __launch_bounds__(256) void k_expand(const half_t* __restrict__ h,
                                                const half_t* __restrict__ star0,
                                                const float* __restrict__ ych,
                                                const half_t* __restrict__ sgate,
                                                const float* __restrict__ blend,
                                                const float* __restrict__ wexp,
                                                const float* __restrict__ bnp,
                                                const float* __restrict__ xin,
                                                float* __restrict__ out) {
  __shared__ alignas(16) half_t smem[2 * 128 * 72];
  half_t* sA = smem;
  half_t* sB = smem + 128 * 72;
  const int tile = blockIdx.x % 50, b = blockIdx.x / 50;
  const int nb = blockIdx.y;               // N half: 0 or 1
  const int hw0 = tile * 128;
  const int tid = threadIdx.x, lane = tid & 31, wv = tid >> 5;
  const int mrow = lane & 15, hi = lane >> 4;
  const float alpha = sigmf(blend[0]);

  v8f acc[8];
#pragma unroll
  for (int i = 0; i < 8; i++)
#pragma unroll
    for (int j = 0; j < 8; j++) acc[i][j] = 0.f;

  for (int kt = 0; kt < 128; kt += 64) {
    for (int i = tid; i < 128 * 64; i += 256) {
      int kl = i >> 7, m = i & 127;
      int ci = kt + kl, cc = ci >> 1;
      float v;
      if (ci & 1) {
        v = (1.f - alpha) * (float)h[((size_t)b * 128 + 64 + cc) * HWN + hw0 + m];
      } else {
        v = alpha * (float)star0[((size_t)b * 64 + cc) * HWN + hw0 + m] *
            ych[b * 64 + cc] * (float)sgate[(size_t)b * HWN + hw0 + m];
      }
      sA[m * 72 + kl] = (half_t)v;
    }
    for (int i = tid; i < 128 * 64; i += 256) {
      int n = i >> 6, kk = i & 63;
      sB[n * 72 + kk] = (half_t)wexp[(size_t)(nb * 128 + n) * 128 + kt + kk];
    }
    __syncthreads();
#pragma unroll
    for (int kk = 0; kk < 64; kk += 32) {
      int koff = kk + hi * 8;
      v16h a = load_frag(sA, wv * 16 + mrow, koff, 72);
      v16h bfr[8];
#pragma unroll
      for (int ns = 0; ns < 8; ns++)
        bfr[ns] = load_frag(sB, ns * 16 + mrow, koff, 72);
#pragma unroll
      for (int ns = 0; ns < 8; ns++)
        acc[ns] = wmma_f16(a, bfr[ns], acc[ns]);
    }
    __syncthreads();
  }

#pragma unroll
  for (int ns = 0; ns < 8; ns++) {
    int ng = nb * 128 + ns * 16 + mrow;
    float s = bnp[ng] * rsqrtf(bnp[768 + ng] + 1e-5f);
    float bi = bnp[256 + ng] - bnp[512 + ng] * s;
#pragma unroll
    for (int r = 0; r < 8; r++) {
      int m = wv * 16 + r + 8 * hi;
      size_t oidx = ((size_t)b * 256 + ng) * HWN + hw0 + m;
      out[oidx] = siluf(acc[ns][r] * s + bi) + xin[oidx];
    }
  }
}

// ---------------------------------------------------------------------------
extern "C" void kernel_launch(void* const* d_in, const int* in_sizes, int n_in,
                              void* d_out, int out_size, void* d_ws, size_t ws_size,
                              hipStream_t stream) {
  const float* x        = (const float*)d_in[0];
  const float* w_reduce = (const float*)d_in[1];
  const float* bn_red   = (const float*)d_in[2];
  const float* off_w    = (const float*)d_in[3];
  const float* off_b    = (const float*)d_in[4];
  const float* dw_w     = (const float*)d_in[5];
  const float* pw_w     = (const float*)d_in[6];
  const float* bn_dcn   = (const float*)d_in[7];
  const float* w_d2     = (const float*)d_in[8];
  const float* bn_d2    = (const float*)d_in[9];
  const float* w_d3     = (const float*)d_in[10];
  const float* bn_d3    = (const float*)d_in[11];
  const float* w_a1     = (const float*)d_in[12];
  const float* bn_a1    = (const float*)d_in[13];
  const float* w_a2     = (const float*)d_in[14];
  const float* bn_a2    = (const float*)d_in[15];
  const float* ctx_w    = (const float*)d_in[16];
  const float* bn_star  = (const float*)d_in[17];
  const float* eca_w    = (const float*)d_in[18];
  const float* sp_w     = (const float*)d_in[19];
  const float* blend_w  = (const float*)d_in[20];
  const float* w_expand = (const float*)d_in[21];
  const float* bn_exp   = (const float*)d_in[22];

  char* ws = (char*)d_ws;
  size_t off = 0;
  auto carve = [&](size_t bytes) -> void* {
    void* p = ws + off;
    off += (bytes + 255) & ~(size_t)255;
    return p;
  };
  const size_t HM16 = (size_t)NBATCH * 64 * HWN * sizeof(half_t);  // 64-ch fp16 map
  half_t* h      = (half_t*)carve(2 * HM16);   // hs || hb, [b][128][hw]
  half_t* xd     = (half_t*)carve(HM16);
  half_t* tbuf   = (half_t*)carve(HM16);
  half_t* dcn    = (half_t*)carve(HM16);
  half_t* a1     = (half_t*)carve(HM16);
  half_t* ctxb   = (half_t*)carve(HM16);
  half_t* star0  = (half_t*)carve(HM16);
  float*  chsum  = (float*)carve(NBATCH * 64 * sizeof(float));
  float*  ych    = (float*)carve(NBATCH * 64 * sizeof(float));
  float*  spmean = (float*)carve((size_t)NBATCH * HWN * sizeof(float));
  float*  spmax  = (float*)carve((size_t)NBATCH * HWN * sizeof(float));
  half_t* sgate  = (half_t*)carve((size_t)NBATCH * HWN * sizeof(half_t));
  (void)in_sizes; (void)n_in; (void)out_size; (void)ws_size;

  k_reduce<<<dim3(NBATCH * 50), 256, 0, stream>>>(x, w_reduce, bn_red, h);
  k_offset<<<dim3(400), 256, 0, stream>>>(h, off_w, off_b, xd);
  k_dw<<<dim3(25600), 256, 0, stream>>>(xd, dw_w, tbuf);
  k_pw<<<dim3(NBATCH * 50), 256, 0, stream>>>(tbuf, pw_w, bn_dcn, dcn);
  k_asym1<<<dim3(25600), 256, 0, stream>>>(h, w_a1, bn_a1, a1);
  k_ctx<<<dim3(25600), 256, 0, stream>>>(h, a1, w_d2, bn_d2, w_d3, bn_d3,
                                         w_a2, bn_a2, ctx_w, ctxb);
  k_star<<<dim3(NBATCH * 64), 256, 0, stream>>>(dcn, ctxb, bn_star, star0, chsum);
  k_eca<<<dim3(1), 1024, 0, stream>>>(chsum, eca_w, ych);
  k_sp<<<dim3(400), 256, 0, stream>>>(star0, ych, spmean, spmax);
  k_gate<<<dim3(400), 256, 0, stream>>>(spmean, spmax, sp_w, sgate);
  k_expand<<<dim3(NBATCH * 50, 2), 256, 0, stream>>>(h, star0, ych, sgate, blend_w,
                                                     w_expand, bn_exp, x,
                                                     (float*)d_out);
}
